// GATNet_19018115186852
// MI455X (gfx1250) — compile-verified, matrix operand order checked by reading, not run on hardware
//
#include <hip/hip_runtime.h>
#include <hip/hip_bf16.h>

typedef __attribute__((ext_vector_type(16))) __bf16 v16bf;
typedef __attribute__((ext_vector_type(8)))  float  v8f;

// ---------------------------------------------------------------------------
// Conversions / init
// ---------------------------------------------------------------------------
__global__ void k_f32_to_bf16(const float* __restrict__ in, __bf16* __restrict__ out,
                              long long n) {
  long long t = (long long)blockIdx.x * blockDim.x + threadIdx.x;
  if (t < n) out[t] = (__bf16)in[t];
}

// W[K][C] f32  ->  Wt[C][K] bf16  (transpose so B-fragment loads match A-layout)
__global__ void k_transpose_to_bf16(const float* __restrict__ W, __bf16* __restrict__ Wt,
                                    int K, int C) {
  long long t = (long long)blockIdx.x * blockDim.x + threadIdx.x;
  if (t >= (long long)K * C) return;
  int k = (int)(t / C);
  int c = (int)(t % C);
  Wt[(long long)c * K + k] = (__bf16)W[t];
}

__global__ void k_zero_f32(float* __restrict__ p, long long n) {
  long long t = (long long)blockIdx.x * blockDim.x + threadIdx.x;
  if (t < n) p[t] = 0.0f;
}

// ---------------------------------------------------------------------------
// bf16 WMMA GEMM: C[M][NC] = A[M][K] * Bt[NC][K]^T   (Bt stored row-major NCxK)
// One wave32 per 16x16 output tile, K-loop in steps of 32 (v_wmma_f32_16x16x32_bf16)
// M and NC must be multiples of 16 (true here: 50000, 256).
// ---------------------------------------------------------------------------
__global__ void k_gemm_bf16(const __bf16* __restrict__ A, const __bf16* __restrict__ Bt,
                            float* __restrict__ C, int M, int K, int NC) {
  int wid  = (int)((blockIdx.x * blockDim.x + threadIdx.x) >> 5);
  int lane = threadIdx.x & 31;
  int ntiles = NC >> 4;
  int total  = (M >> 4) * ntiles;
  if (wid >= total) return;                 // wave-uniform branch (EXEC stays all-1)
  int mt = wid / ntiles;
  int nt = wid % ntiles;

  int r  = lane & 15;                       // row (A) / col (B) within tile
  int kb = (lane >> 4) << 3;                // K sub-offset: 0 or 8

  const __bf16* arow = A  + (long long)(mt * 16 + r) * K;
  const __bf16* brow = Bt + (long long)(nt * 16 + r) * K;

  v8f acc = {};
  for (int k0 = 0; k0 < K; k0 += 32) {
    v16bf a, b;
#pragma unroll
    for (int v = 0; v < 4; ++v) {
      int k = k0 + kb + 2 * v;
      a[2 * v]       = arow[k];
      a[2 * v + 1]   = arow[k + 1];
      a[8 + 2 * v]     = arow[k + 16];
      a[8 + 2 * v + 1] = arow[k + 17];
      b[2 * v]       = brow[k];
      b[2 * v + 1]   = brow[k + 1];
      b[8 + 2 * v]     = brow[k + 16];
      b[8 + 2 * v + 1] = brow[k + 17];
    }
    acc = __builtin_amdgcn_wmma_f32_16x16x32_bf16(false, a, false, b,
                                                  (short)0, acc, false, false);
  }

  int col     = nt * 16 + r;
  int rowbase = mt * 16 + ((lane >> 4) << 3);   // lanes 16-31 hold M = r+8
#pragma unroll
  for (int i = 0; i < 8; ++i)
    C[(long long)(rowbase + i) * NC + col] = acc[i];
}

// ---------------------------------------------------------------------------
// Edge phase (shared by both layers: stride 256 = 8 heads * 32 dims)
// Self-loops implicit: eid >= E  ->  src = dst = eid - E
// ---------------------------------------------------------------------------
__device__ __forceinline__ void edge_nodes(const long long* src64, const long long* dst64,
                                           long long eid, int E, int& s, int& d) {
  if (eid < E) { s = (int)src64[eid]; d = (int)dst64[eid]; }
  else         { s = d = (int)(eid - E); }
}

__global__ void k_edge_logits(const float* __restrict__ XL, const float* __restrict__ XR,
                              const float* __restrict__ att,
                              const long long* __restrict__ src64,
                              const long long* __restrict__ dst64,
                              int E, int N, float* __restrict__ e_out,
                              unsigned int* __restrict__ m_enc) {
  long long t = (long long)blockIdx.x * blockDim.x + threadIdx.x;
  long long total = ((long long)E + N) * 8;
  if (t >= total) return;
  int h = (int)(t & 7);
  long long eid = t >> 3;
  int s, d;
  edge_nodes(src64, dst64, eid, E, s, d);

  const float* a  = att + h * 32;
  const float* xl = XL + (long long)s * 256 + h * 32;
  const float* xr = XR + (long long)d * 256 + h * 32;

  float acc = 0.0f;
#pragma unroll
  for (int i = 0; i < 32; i += 4) {
    float4 u = *(const float4*)(xl + i);
    float4 v = *(const float4*)(xr + i);
    float4 w = *(const float4*)(a + i);
    float z;
    z = u.x + v.x; z = z > 0.0f ? z : 0.2f * z; acc += z * w.x;
    z = u.y + v.y; z = z > 0.0f ? z : 0.2f * z; acc += z * w.y;
    z = u.z + v.z; z = z > 0.0f ? z : 0.2f * z; acc += z * w.z;
    z = u.w + v.w; z = z > 0.0f ? z : 0.2f * z; acc += z * w.w;
  }
  e_out[t] = acc;

  unsigned int bits = __float_as_uint(acc);
  unsigned int enc  = (bits & 0x80000000u) ? ~bits : (bits | 0x80000000u);
  atomicMax(&m_enc[(long long)d * 8 + h], enc);
}

__global__ void k_edge_exp(float* __restrict__ e_io,
                           const unsigned int* __restrict__ m_enc,
                           float* __restrict__ S,
                           const long long* __restrict__ src64,
                           const long long* __restrict__ dst64,
                           int E, int N) {
  long long t = (long long)blockIdx.x * blockDim.x + threadIdx.x;
  long long total = ((long long)E + N) * 8;
  if (t >= total) return;
  int h = (int)(t & 7);
  long long eid = t >> 3;
  int s, d;
  edge_nodes(src64, dst64, eid, E, s, d);
  (void)s;

  unsigned int enc  = m_enc[(long long)d * 8 + h];
  unsigned int bits = (enc & 0x80000000u) ? (enc & 0x7FFFFFFFu) : ~enc;
  float m = __uint_as_float(bits);
  float ex = expf(e_io[t] - m);
  e_io[t] = ex;
  atomicAdd(&S[(long long)d * 8 + h], ex);
}

__global__ void k_edge_agg(const float* __restrict__ ex,
                           const float* __restrict__ S,
                           const float* __restrict__ XL,
                           float* __restrict__ AGG,
                           const long long* __restrict__ src64,
                           const long long* __restrict__ dst64,
                           int E, int N) {
  long long t = (long long)blockIdx.x * blockDim.x + threadIdx.x;
  long long total = ((long long)E + N) * 8;
  if (t >= total) return;
  int h = (int)(t & 7);
  long long eid = t >> 3;
  int s, d;
  edge_nodes(src64, dst64, eid, E, s, d);

  float alpha = ex[t] / (S[(long long)d * 8 + h] + 1e-16f);
  const float* xl = XL + (long long)s * 256 + h * 32;
  float* o = AGG + (long long)d * 256 + h * 32;
#pragma unroll
  for (int i = 0; i < 32; ++i) atomicAdd(o + i, alpha * xl[i]);
}

// layer1 epilogue: h = ELU(AGG + b1), stored as bf16 for layer2 GEMM
__global__ void k_finalize1(const float* __restrict__ AGG, const float* __restrict__ b1,
                            __bf16* __restrict__ hb, long long n) {
  long long t = (long long)blockIdx.x * blockDim.x + threadIdx.x;
  if (t >= n) return;
  float v = AGG[t] + b1[t & 255];
  v = v > 0.0f ? v : expm1f(v);
  hb[t] = (__bf16)v;
}

// layer2 epilogue: out[n][d] = mean_h AGG[n][h*32+d] + b2[d]
__global__ void k_finalize2(const float* __restrict__ AGG, const float* __restrict__ b2,
                            float* __restrict__ out, int N) {
  long long t = (long long)blockIdx.x * blockDim.x + threadIdx.x;
  if (t >= (long long)N * 32) return;
  int n = (int)(t >> 5);
  int d = (int)(t & 31);
  const float* base = AGG + (long long)n * 256 + d;
  float acc = 0.0f;
#pragma unroll
  for (int h = 0; h < 8; ++h) acc += base[h * 32];
  out[t] = acc * 0.125f + b2[d];
}

// ---------------------------------------------------------------------------
// Host driver
// ---------------------------------------------------------------------------
static inline int blocks_for(long long n, int bs) { return (int)((n + bs - 1) / bs); }

extern "C" void kernel_launch(void* const* d_in, const int* in_sizes, int n_in,
                              void* d_out, int out_size, void* d_ws, size_t ws_size,
                              hipStream_t stream) {
  (void)n_in; (void)out_size; (void)ws_size;

  const float*     x    = (const float*)d_in[0];
  const long long* ei   = (const long long*)d_in[1];   // int64 edge_index [2][E]
  const float*     Wl1  = (const float*)d_in[2];
  const float*     Wr1  = (const float*)d_in[3];
  const float*     att1 = (const float*)d_in[4];
  const float*     b1   = (const float*)d_in[5];
  const float*     Wl2  = (const float*)d_in[6];
  const float*     Wr2  = (const float*)d_in[7];
  const float*     att2 = (const float*)d_in[8];
  const float*     b2   = (const float*)d_in[9];
  float*           out  = (float*)d_out;

  const int DIN = 128, C1 = 256, H = 8;
  const int N = in_sizes[0] / DIN;
  const int E = in_sizes[1] / 2;
  const long long ET = (long long)E + N;               // edges incl. self-loops

  // workspace layout (256B aligned slices)
  char* w = (char*)d_ws;
  size_t off = 0;
  auto take = [&](size_t bytes) {
    size_t o = off;
    off = (off + bytes + 255) & ~(size_t)255;
    return o;
  };
  __bf16*       ABF  = (__bf16*)(w + take((size_t)N * C1 * sizeof(__bf16)));
  __bf16*       WLT  = (__bf16*)(w + take((size_t)C1 * C1 * sizeof(__bf16)));
  __bf16*       WRT  = (__bf16*)(w + take((size_t)C1 * C1 * sizeof(__bf16)));
  float*        XL   = (float*)(w + take((size_t)N * C1 * sizeof(float)));
  float*        XR   = (float*)(w + take((size_t)N * C1 * sizeof(float)));
  float*        AGG  = (float*)(w + take((size_t)N * C1 * sizeof(float)));
  float*        ED   = (float*)(w + take((size_t)ET * H * sizeof(float)));
  unsigned int* MEnc = (unsigned int*)(w + take((size_t)N * H * sizeof(unsigned int)));
  float*        S    = (float*)(w + take((size_t)N * H * sizeof(float)));

  const long long* src64 = ei;
  const long long* dst64 = ei + E;

  const int BS = 256;
  const int gemm_waves = (N / 16) * (C1 / 16);
  const int gemm_blocks = blocks_for((long long)gemm_waves * 32, BS);
  const long long nEdgeT = ET * H;

  // ---------------- Layer 1 ----------------
  {
    long long nx = (long long)N * DIN;
    k_f32_to_bf16<<<blocks_for(nx, BS), BS, 0, stream>>>(x, ABF, nx);
    k_transpose_to_bf16<<<blocks_for((long long)DIN * C1, BS), BS, 0, stream>>>(Wl1, WLT, DIN, C1);
    k_transpose_to_bf16<<<blocks_for((long long)DIN * C1, BS), BS, 0, stream>>>(Wr1, WRT, DIN, C1);

    k_gemm_bf16<<<gemm_blocks, BS, 0, stream>>>(ABF, WLT, XL, N, DIN, C1);
    k_gemm_bf16<<<gemm_blocks, BS, 0, stream>>>(ABF, WRT, XR, N, DIN, C1);

    k_zero_f32<<<blocks_for((long long)N * H, BS), BS, 0, stream>>>((float*)MEnc, (long long)N * H);
    k_zero_f32<<<blocks_for((long long)N * H, BS), BS, 0, stream>>>(S, (long long)N * H);
    k_zero_f32<<<blocks_for((long long)N * C1, BS), BS, 0, stream>>>(AGG, (long long)N * C1);

    k_edge_logits<<<blocks_for(nEdgeT, BS), BS, 0, stream>>>(XL, XR, att1, src64, dst64, E, N, ED, MEnc);
    k_edge_exp<<<blocks_for(nEdgeT, BS), BS, 0, stream>>>(ED, MEnc, S, src64, dst64, E, N);
    k_edge_agg<<<blocks_for(nEdgeT, BS), BS, 0, stream>>>(ED, S, XL, AGG, src64, dst64, E, N);

    k_finalize1<<<blocks_for((long long)N * C1, BS), BS, 0, stream>>>(AGG, b1, ABF, (long long)N * C1);
  }

  // ---------------- Layer 2 ----------------
  {
    k_transpose_to_bf16<<<blocks_for((long long)C1 * C1, BS), BS, 0, stream>>>(Wl2, WLT, C1, C1);
    k_transpose_to_bf16<<<blocks_for((long long)C1 * C1, BS), BS, 0, stream>>>(Wr2, WRT, C1, C1);

    k_gemm_bf16<<<gemm_blocks, BS, 0, stream>>>(ABF, WLT, XL, N, C1, C1);
    k_gemm_bf16<<<gemm_blocks, BS, 0, stream>>>(ABF, WRT, XR, N, C1, C1);

    k_zero_f32<<<blocks_for((long long)N * H, BS), BS, 0, stream>>>((float*)MEnc, (long long)N * H);
    k_zero_f32<<<blocks_for((long long)N * H, BS), BS, 0, stream>>>(S, (long long)N * H);
    k_zero_f32<<<blocks_for((long long)N * C1, BS), BS, 0, stream>>>(AGG, (long long)N * C1);

    k_edge_logits<<<blocks_for(nEdgeT, BS), BS, 0, stream>>>(XL, XR, att2, src64, dst64, E, N, ED, MEnc);
    k_edge_exp<<<blocks_for(nEdgeT, BS), BS, 0, stream>>>(ED, MEnc, S, src64, dst64, E, N);
    k_edge_agg<<<blocks_for(nEdgeT, BS), BS, 0, stream>>>(ED, S, XL, AGG, src64, dst64, E, N);

    k_finalize2<<<blocks_for((long long)N * 32, BS), BS, 0, stream>>>(AGG, b2, out, N);
  }
}
